// DualStreamDTIClassifier_43224550867588
// MI455X (gfx1250) — matrix operand order repaired and checked by reading, not compiled, not run on hardware
//
#include <hip/hip_runtime.h>
#include <hip/hip_bf16.h>
#include <math.h>

typedef __bf16 bf16_t;
typedef __attribute__((ext_vector_type(16))) __bf16 v16bf;
typedef __attribute__((ext_vector_type(8)))  float  v8f;

#define B_SZ 8
#define LQ   256
#define LK   1024
#define DM   1024
#define NH   16
#define DH   64
#define DFF  4096
#define NL   4

#define BLK_M 128
#define BLK_N 128
#define KTILE 32
#define LDSP  40   // LDS row pitch in bf16 elems (80B, 16B-aligned, conflict padding)

union Frag { v16bf v; uint4 u[2]; };

// ---------------------------------------------------------------------------
// Tiled WMMA GEMM:  C[M,N] (+)= A[M,K](bf16,row-major) * Bt[N,K](bf16,row-major)^T + bias
// EPI: 0 = fp32 store, 1 = bf16(acc+bias), 2 = bf16(gelu(acc+bias)), 3 = fp32 C += acc+bias
// Batched over blockIdx.z with composite C offset (outer/inner strides).
//
// Staging: GLOBAL_LOAD_ASYNC_TO_LDS_B128 (memory->LDS, no VGPR round trip,
// ASYNCcnt-tracked) into double-buffered LDS; WMMAs on tile k overlap the
// async fill of tile k+1. Out-of-range rows are CLAMPED (not zero-filled):
// the resulting garbage accumulators are never stored, and this keeps the
// staging path branch-free (EXEC stays all-ones for WMMA).
// ---------------------------------------------------------------------------
template<int EPI>
__global__ __launch_bounds__(256) void gemm_bf16_kernel(
    const bf16_t* __restrict__ A, const bf16_t* __restrict__ Bt,
    void* __restrict__ Cv, const float* __restrict__ bias,
    int M, int N, int K, int ldc,
    long long strideA, long long strideB,
    long long strideCo, long long strideCi, int innerCnt)
{
  __shared__ bf16_t As[2][BLK_M * LDSP];
  __shared__ bf16_t Bs[2][BLK_N * LDSP];

  const int z = blockIdx.z;
  A  += (size_t)z * strideA;
  Bt += (size_t)z * strideB;
  const long long coff = (long long)(z / innerCnt) * strideCo
                       + (long long)(z % innerCnt) * strideCi;

  const int m0   = blockIdx.y * BLK_M;
  const int n0   = blockIdx.x * BLK_N;
  const int tid  = threadIdx.x;
  const int lane = tid & 31;
  const int wid  = tid >> 5;
  const int waveM = wid >> 1;   // 0..3 -> 32-row strip
  const int waveN = wid & 1;    // 0..1 -> 64-col strip
  const int half = lane >> 4;
  const int lm   = lane & 15;

  // staging coordinates: 128 rows x 4 16B-chunks, 2 rounds of 256 threads
  const int srow = tid >> 2;          // 0..63
  const int skc  = (tid & 3) << 3;    // 0,8,16,24 (bf16 elems)

  const unsigned ldsA = (unsigned)(uintptr_t)(&As[0][0]);
  const unsigned ldsB = (unsigned)(uintptr_t)(&Bs[0][0]);
  const unsigned bufBytesA = (unsigned)(BLK_M * LDSP * 2);
  const unsigned bufBytesB = (unsigned)(BLK_N * LDSP * 2);

  auto stage = [&](int buf, int kk) {
    #pragma unroll
    for (int i = 0; i < 2; ++i) {
      int row = srow + i * 64;
      {
        int gm = m0 + row; gm = gm < M ? gm : M - 1;            // clamp, no branch
        unsigned long long ga = (unsigned long long)(A + (size_t)gm * K + (kk + skc));
        unsigned la = ldsA + (unsigned)buf * bufBytesA + (unsigned)((row * LDSP + skc) * 2);
        asm volatile("global_load_async_to_lds_b128 %0, %1, off"
                     :: "v"(la), "v"(ga) : "memory");
      }
      {
        int gn = n0 + row; gn = gn < N ? gn : N - 1;
        unsigned long long gb = (unsigned long long)(Bt + (size_t)gn * K + (kk + skc));
        unsigned lb = ldsB + (unsigned)buf * bufBytesB + (unsigned)((row * LDSP + skc) * 2);
        asm volatile("global_load_async_to_lds_b128 %0, %1, off"
                     :: "v"(lb), "v"(gb) : "memory");
      }
    }
  };

  v8f acc[2][4];
  {
    v8f zero = {0.f,0.f,0.f,0.f,0.f,0.f,0.f,0.f};
    #pragma unroll
    for (int i = 0; i < 2; ++i)
      #pragma unroll
      for (int j = 0; j < 4; ++j) acc[i][j] = zero;
  }

  stage(0, 0);                       // prologue fill of buffer 0

  int buf = 0;
  for (int k0 = 0; k0 < K; k0 += KTILE) {
    // my async fills of `buf` (issued last iteration or prologue) must land,
    // then barrier so every wave's fills are visible.
    asm volatile("s_wait_asynccnt 0x0" ::: "memory");
    __syncthreads();

    if (k0 + KTILE < K) stage(buf ^ 1, k0 + KTILE);   // overlap next tile with WMMA

    // ---- fragment loads per documented CDNA5 wave32 layouts ----
    Frag af[2], bfr[4];
    const bf16_t* Ab = &As[buf][0];
    const bf16_t* Bb = &Bs[buf][0];
    #pragma unroll
    for (int fm = 0; fm < 2; ++fm) {
      const bf16_t* p = Ab + (waveM * 32 + fm * 16 + lm) * LDSP;
      af[fm].u[0] = *(const uint4*)(p + half * 8);        // k = half*8 + j   (j=0..7)
      af[fm].u[1] = *(const uint4*)(p + 16 + half * 8);   // k = 16+half*8+.. (j=8..15)
    }
    #pragma unroll
    for (int fn = 0; fn < 4; ++fn) {
      const bf16_t* p = Bb + (waveN * 64 + fn * 16 + lm) * LDSP;
      bfr[fn].u[0] = *(const uint4*)(p + half * 8);       // k = half*8 + j   (j=0..7)
      bfr[fn].u[1] = *(const uint4*)(p + 16 + half * 8);
    }
    #pragma unroll
    for (int fm = 0; fm < 2; ++fm)
      #pragma unroll
      for (int fn = 0; fn < 4; ++fn)
        acc[fm][fn] = __builtin_amdgcn_wmma_f32_16x16x32_bf16(
            false, af[fm].v, false, bfr[fn].v, (short)0, acc[fm][fn], false, false);

    buf ^= 1;
  }

  // ---- epilogue (C layout: lane n = lm, row m = half*8 + r) ----
  #pragma unroll
  for (int fm = 0; fm < 2; ++fm) {
    #pragma unroll
    for (int fn = 0; fn < 4; ++fn) {
      int n = n0 + waveN * 64 + fn * 16 + lm;
      if (n >= N) continue;
      float bv = bias ? bias[n] : 0.0f;
      #pragma unroll
      for (int r = 0; r < 8; ++r) {
        int m = m0 + waveM * 32 + fm * 16 + half * 8 + r;
        if (m >= M) continue;
        float v = acc[fm][fn][r] + bv;
        size_t ci = (size_t)coff + (size_t)m * ldc + n;
        if constexpr (EPI == 0) {
          ((float*)Cv)[ci] = v;
        } else if constexpr (EPI == 1) {
          ((bf16_t*)Cv)[ci] = (bf16_t)v;
        } else if constexpr (EPI == 2) {
          float g = 0.5f * v * (1.0f + erff(v * 0.70710678118654752f));
          ((bf16_t*)Cv)[ci] = (bf16_t)g;
        } else {
          ((float*)Cv)[ci] += v;
        }
      }
    }
  }
}

// ---------------------------------------------------------------------------
// LayerNorm (fp32 in, bf16 out), one 256-thread block per row of 1024
// ---------------------------------------------------------------------------
__global__ __launch_bounds__(256) void ln_kernel(
    const float* __restrict__ x, const float* __restrict__ g,
    const float* __restrict__ b, bf16_t* __restrict__ out)
{
  const int row = blockIdx.x;
  const float* xr = x + (size_t)row * DM;
  float vals[4], s = 0.f, s2 = 0.f;
  #pragma unroll
  for (int j = 0; j < 4; ++j) {
    float v = xr[threadIdx.x + j * 256];
    vals[j] = v; s += v; s2 += v * v;
  }
  __shared__ float sh1[8], sh2[8];
  for (int o = 16; o; o >>= 1) { s += __shfl_xor(s, o, 32); s2 += __shfl_xor(s2, o, 32); }
  int wid = threadIdx.x >> 5, lane = threadIdx.x & 31;
  if (!lane) { sh1[wid] = s; sh2[wid] = s2; }
  __syncthreads();
  if (wid == 0) {
    float a = lane < 8 ? sh1[lane] : 0.f;
    float c = lane < 8 ? sh2[lane] : 0.f;
    for (int o = 4; o; o >>= 1) { a += __shfl_xor(a, o, 32); c += __shfl_xor(c, o, 32); }
    if (!lane) { sh1[0] = a; sh2[0] = c; }
  }
  __syncthreads();
  float mean = sh1[0] * (1.f / DM);
  float var  = sh2[0] * (1.f / DM) - mean * mean;
  float rstd = rsqrtf(var + 1e-5f);
  #pragma unroll
  for (int j = 0; j < 4; ++j) {
    int c = threadIdx.x + j * 256;
    out[(size_t)row * DM + c] = (bf16_t)((vals[j] - mean) * rstd * g[c] + b[c]);
  }
}

// ---------------------------------------------------------------------------
// L2-normalize per (b,l,h) over dh=64 and rearrange (B,L,H,dh) -> (B,H,L,dh).
// One wave per (b,l,h); mult folds attention scale into q.
// ---------------------------------------------------------------------------
__global__ __launch_bounds__(256) void l2norm_arrange_kernel(
    const bf16_t* __restrict__ in, bf16_t* __restrict__ out, int Lseq, float mult)
{
  int wgid = blockIdx.x * 8 + (threadIdx.x >> 5);
  int lane = threadIdx.x & 31;
  int h  = wgid & 15;
  int bl = wgid >> 4;              // b*Lseq + l
  size_t ibase = (size_t)bl * DM + h * DH;
  float v0 = (float)in[ibase + lane];
  float v1 = (float)in[ibase + lane + 32];
  float ss = v0 * v0 + v1 * v1;
  for (int o = 16; o; o >>= 1) ss += __shfl_xor(ss, o, 32);
  float inv = mult / fmaxf(sqrtf(ss), 1e-12f);
  int b = bl / Lseq, l = bl % Lseq;
  size_t obase = (((size_t)(b * NH + h) * Lseq) + l) * DH;
  out[obase + lane]      = (bf16_t)(v0 * inv);
  out[obase + lane + 32] = (bf16_t)(v1 * inv);
}

// (B,LK,NH,DH) -> (B,NH,DH,LK)   so ctx GEMM reads V^T rows contiguously
__global__ __launch_bounds__(256) void vtrans_kernel(
    const bf16_t* __restrict__ in, bf16_t* __restrict__ out)
{
  int idx = blockIdx.x * 256 + threadIdx.x;      // B*LK*DM elems
  int d = idx & 63;
  int h = (idx >> 6) & 15;
  int l = (idx >> 10) & 1023;
  int b = idx >> 20;
  out[((((size_t)(b * NH + h)) * DH + d) << 10) + l] = in[idx];
}

// ---------------------------------------------------------------------------
// Masked softmax + renorm + eps-smoothing; one block per (b,h,q) row of LK
// ---------------------------------------------------------------------------
__global__ __launch_bounds__(256) void softmax_kernel(
    const float* __restrict__ logits, const unsigned char* __restrict__ pad,
    bf16_t* __restrict__ attn)
{
  const int row = blockIdx.x;                // b*NH*LQ + h*LQ + q
  const int b = row / (NH * LQ);
  const float* lr = logits + (size_t)row * LK;
  const unsigned char* pr = pad + (size_t)b * LK;

  float lv[4]; int vd[4];
  float mx = -1e30f, cnt = 0.f;
  #pragma unroll
  for (int j = 0; j < 4; ++j) {
    int k = threadIdx.x + j * 256;
    vd[j] = (pr[k] == 0);
    lv[j] = lr[k];
    if (vd[j]) { mx = fmaxf(mx, lv[j]); cnt += 1.f; }
  }
  __shared__ float shm[8], shc[8], shs[8];
  int wid = threadIdx.x >> 5, lane = threadIdx.x & 31;
  for (int o = 16; o; o >>= 1) { mx = fmaxf(mx, __shfl_xor(mx, o, 32)); cnt += __shfl_xor(cnt, o, 32); }
  if (!lane) { shm[wid] = mx; shc[wid] = cnt; }
  __syncthreads();
  if (wid == 0) {
    float a = lane < 8 ? shm[lane] : -1e30f;
    float c = lane < 8 ? shc[lane] : 0.f;
    for (int o = 4; o; o >>= 1) { a = fmaxf(a, __shfl_xor(a, o, 32)); c += __shfl_xor(c, o, 32); }
    if (!lane) { shm[0] = a; shc[0] = c; }
  }
  __syncthreads();
  mx = shm[0]; cnt = shc[0];

  float e[4], s = 0.f;
  #pragma unroll
  for (int j = 0; j < 4; ++j) { e[j] = vd[j] ? expf(lv[j] - mx) : 0.f; s += e[j]; }
  for (int o = 16; o; o >>= 1) s += __shfl_xor(s, o, 32);
  if (!lane) shs[wid] = s;
  __syncthreads();
  if (wid == 0) {
    float a = lane < 8 ? shs[lane] : 0.f;
    for (int o = 4; o; o >>= 1) a += __shfl_xor(a, o, 32);
    if (!lane) shs[0] = a;
  }
  __syncthreads();
  s = shs[0];

  float inv  = (1.0f - 0.1f) / fmaxf(s, 1e-8f);
  float unif = 0.1f / fmaxf(cnt, 1.f);
  #pragma unroll
  for (int j = 0; j < 4; ++j) {
    int k = threadIdx.x + j * 256;
    float a = vd[j] ? (e[j] * inv + unif) : 0.f;
    attn[(size_t)row * LK + k] = (bf16_t)a;
  }
}

// fp32 (K,N) row-major -> bf16 (N,K) row-major, per-layer batch in z
__global__ __launch_bounds__(256) void transpose_cvt_kernel(
    const float* __restrict__ in, bf16_t* __restrict__ out, int K, int N)
{
  __shared__ float tile[32][33];
  size_t zoff = (size_t)blockIdx.z * K * N;
  int n0 = blockIdx.x * 32, k0 = blockIdx.y * 32;
  int tx = threadIdx.x & 31, ty = threadIdx.x >> 5;  // ty 0..7
  #pragma unroll
  for (int j = 0; j < 4; ++j) {
    int k = k0 + ty + j * 8, n = n0 + tx;
    if (k < K && n < N) tile[ty + j * 8][tx] = in[zoff + (size_t)k * N + n];
  }
  __syncthreads();
  #pragma unroll
  for (int j = 0; j < 4; ++j) {
    int n = n0 + ty + j * 8, k = k0 + tx;
    if (n < N && k < K) out[zoff + (size_t)n * K + k] = (bf16_t)tile[tx][ty + j * 8];
  }
}

__global__ __launch_bounds__(256) void copy_f32_kernel(
    const float* __restrict__ in, float* __restrict__ out, int n)
{
  int i = blockIdx.x * 256 + threadIdx.x;
  if (i < n) out[i] = in[i];
}

// ---------------------------------------------------------------------------
static inline int cdiv(int a, int b) { return (a + b - 1) / b; }

template<int EPI>
static void launch_gemm(hipStream_t s, const bf16_t* A, const bf16_t* Bt, void* C,
                        const float* bias, int M, int N, int K, int ldc,
                        long long sA, long long sB, long long sCo, long long sCi,
                        int innerCnt, int batches)
{
  dim3 grid(cdiv(N, BLK_N), cdiv(M, BLK_M), batches);
  gemm_bf16_kernel<EPI><<<grid, dim3(256), 0, s>>>(
      A, Bt, C, bias, M, N, K, ldc, sA, sB, sCo, sCi, innerCnt);
}

extern "C" void kernel_launch(void* const* d_in, const int* in_sizes, int n_in,
                              void* d_out, int out_size, void* d_ws, size_t ws_size,
                              hipStream_t stream)
{
  (void)in_sizes; (void)n_in; (void)out_size; (void)ws_size;
  const float* l_h   = (const float*)d_in[0];
  const float* p_h   = (const float*)d_in[1];
  const unsigned char* p_pad = (const unsigned char*)d_in[2];
  const float* ln_q_g  = (const float*)d_in[3];
  const float* ln_q_b  = (const float*)d_in[4];
  const float* ln_kv_g = (const float*)d_in[5];
  const float* ln_kv_b = (const float*)d_in[6];
  const float* ln_f_g  = (const float*)d_in[7];
  const float* ln_f_b  = (const float*)d_in[8];
  const float* Wq = (const float*)d_in[9];
  const float* bq = (const float*)d_in[10];
  const float* Wk = (const float*)d_in[11];
  const float* bk = (const float*)d_in[12];
  const float* Wv = (const float*)d_in[13];
  const float* bv = (const float*)d_in[14];
  const float* Wo = (const float*)d_in[15];
  const float* bo = (const float*)d_in[16];
  const float* W1 = (const float*)d_in[17];
  const float* b1 = (const float*)d_in[18];
  const float* W2 = (const float*)d_in[19];
  const float* b2 = (const float*)d_in[20];
  float* out = (float*)d_out;     // used as fp32 residual stream l_h

  char* ws = (char*)d_ws;
  size_t off = 0;
  auto walloc = [&](size_t bytes) -> char* {
    char* p = ws + off; off += (bytes + 255) & ~(size_t)255; return p;
  };
  bf16_t* wqT = (bf16_t*)walloc((size_t)NL * DM * DM * 2);
  bf16_t* wkT = (bf16_t*)walloc((size_t)NL * DM * DM * 2);
  bf16_t* wvT = (bf16_t*)walloc((size_t)NL * DM * DM * 2);
  bf16_t* woT = (bf16_t*)walloc((size_t)NL * DM * DM * 2);
  bf16_t* w1T = (bf16_t*)walloc((size_t)NL * DM * DFF * 2);
  bf16_t* w2T = (bf16_t*)walloc((size_t)NL * DFF * DM * 2);
  bf16_t* lq   = (bf16_t*)walloc((size_t)B_SZ * LQ * DM * 2);
  bf16_t* pkv  = (bf16_t*)walloc((size_t)B_SZ * LK * DM * 2);
  bf16_t* qtmp = (bf16_t*)walloc((size_t)B_SZ * LQ * DM * 2);
  bf16_t* ktmp = (bf16_t*)walloc((size_t)B_SZ * LK * DM * 2);
  bf16_t* vtmp = (bf16_t*)walloc((size_t)B_SZ * LK * DM * 2);
  bf16_t* qarr = (bf16_t*)walloc((size_t)B_SZ * LQ * DM * 2);
  bf16_t* karr = (bf16_t*)walloc((size_t)B_SZ * LK * DM * 2);
  bf16_t* vt   = (bf16_t*)walloc((size_t)B_SZ * LK * DM * 2);
  float*  logits = (float*)walloc((size_t)B_SZ * NH * LQ * LK * 4);
  bf16_t* attn = (bf16_t*)walloc((size_t)B_SZ * NH * LQ * LK * 2);
  bf16_t* ctx  = (bf16_t*)walloc((size_t)B_SZ * LQ * DM * 2);
  bf16_t* xbuf   = qtmp;   // disjoint lifetime within a layer
  bf16_t* hidden = ktmp;   // disjoint lifetime within a layer (16MB each)

  // ---- one-time (per launch) weight convert + transpose to bf16 ----
  dim3 tb(256);
  transpose_cvt_kernel<<<dim3(DM / 32, DM / 32, NL), tb, 0, stream>>>(Wq, wqT, DM, DM);
  transpose_cvt_kernel<<<dim3(DM / 32, DM / 32, NL), tb, 0, stream>>>(Wk, wkT, DM, DM);
  transpose_cvt_kernel<<<dim3(DM / 32, DM / 32, NL), tb, 0, stream>>>(Wv, wvT, DM, DM);
  transpose_cvt_kernel<<<dim3(DM / 32, DM / 32, NL), tb, 0, stream>>>(Wo, woT, DM, DM);
  transpose_cvt_kernel<<<dim3(DFF / 32, DM / 32, NL), tb, 0, stream>>>(W1, w1T, DM, DFF);
  transpose_cvt_kernel<<<dim3(DM / 32, DFF / 32, NL), tb, 0, stream>>>(W2, w2T, DFF, DM);

  copy_f32_kernel<<<(B_SZ * LQ * DM) / 256, tb, 0, stream>>>(l_h, out, B_SZ * LQ * DM);

  const float attn_scale = 0.125f;  // 1/sqrt(dh)/ATTN_TEMP, folded into q

  for (int i = 0; i < NL; ++i) {
    const bf16_t* wq_i = wqT + (size_t)i * DM * DM;
    const bf16_t* wk_i = wkT + (size_t)i * DM * DM;
    const bf16_t* wv_i = wvT + (size_t)i * DM * DM;
    const bf16_t* wo_i = woT + (size_t)i * DM * DM;
    const bf16_t* w1_i = w1T + (size_t)i * DM * DFF;
    const bf16_t* w2_i = w2T + (size_t)i * DFF * DM;

    // LN(q-stream), LN(kv-stream)
    ln_kernel<<<B_SZ * LQ, tb, 0, stream>>>(out, ln_q_g + i * DM, ln_q_b + i * DM, lq);
    ln_kernel<<<B_SZ * LK, tb, 0, stream>>>(p_h, ln_kv_g + i * DM, ln_kv_b + i * DM, pkv);

    // QKV projections (bf16 out with bias)
    launch_gemm<1>(stream, lq,  wq_i, qtmp, bq + i * DM, B_SZ * LQ, DM, DM, DM, 0, 0, 0, 0, 1, 1);
    launch_gemm<1>(stream, pkv, wk_i, ktmp, bk + i * DM, B_SZ * LK, DM, DM, DM, 0, 0, 0, 0, 1, 1);
    launch_gemm<1>(stream, pkv, wv_i, vtmp, bv + i * DM, B_SZ * LK, DM, DM, DM, 0, 0, 0, 0, 1, 1);

    // l2norm + head rearrange (scale folded into q); V transposed to (B,H,dh,Lk)
    l2norm_arrange_kernel<<<(B_SZ * LQ * NH) / 8, tb, 0, stream>>>(qtmp, qarr, LQ, attn_scale);
    l2norm_arrange_kernel<<<(B_SZ * LK * NH) / 8, tb, 0, stream>>>(ktmp, karr, LK, 1.0f);
    vtrans_kernel<<<(B_SZ * LK * DM) / 256, tb, 0, stream>>>(vtmp, vt);

    // logits = q @ k^T  (batched over B*H; k rows ARE Bt rows)  fp32 out
    launch_gemm<0>(stream, qarr, karr, logits, nullptr, LQ, LK, DH, LK,
                   (long long)LQ * DH, (long long)LK * DH,
                   (long long)LQ * LK, 0, 1, B_SZ * NH);

    // masked softmax + smoothing -> bf16 attn
    softmax_kernel<<<B_SZ * NH * LQ, tb, 0, stream>>>(logits, p_pad, attn);

    // ctx = attn @ v  (batched; Bt = V^T rows)  -> (B,Lq,H*dh) bf16
    launch_gemm<1>(stream, attn, vt, ctx, nullptr, LQ, DH, LK, DM,
                   (long long)LQ * LK, (long long)DH * LK,
                   (long long)LQ * DM, (long long)DH, NH, B_SZ * NH);

    // l_h += ctx @ Wo + bo   (fp32 residual accumulate into d_out)
    launch_gemm<3>(stream, ctx, wo_i, out, bo + i * DM, B_SZ * LQ, DM, DM, DM, 0, 0, 0, 0, 1, 1);

    // FFN: x = LN(l_h); h = gelu(x@W1+b1); l_h += h@W2+b2
    ln_kernel<<<B_SZ * LQ, tb, 0, stream>>>(out, ln_f_g + i * DM, ln_f_b + i * DM, xbuf);
    launch_gemm<2>(stream, xbuf, w1_i, hidden, b1 + i * DFF, B_SZ * LQ, DFF, DM, DFF, 0, 0, 0, 0, 1, 1);
    launch_gemm<3>(stream, hidden, w2_i, out, b2 + i * DM, B_SZ * LQ, DM, DFF, DM, 0, 0, 0, 0, 1, 1);
  }
}
